// GraphRCNN_83889301225994
// MI455X (gfx1250) — matrix operand order repaired and checked by reading, not compiled, not run on hardware
//
#include <hip/hip_runtime.h>
#include <hip/hip_bf16.h>

// ---------------------------------------------------------------------------
// RGCN forward for MI455X (gfx1250, wave32, WMMA).
// Aggregate-first for every layer (mean commutes with the linear map):
// per layer = scatter (edge atomics) + two bf16-WMMA GEMMs with fused
// bias/ReLU. GEMM stages tiles into LDS in *fragment-packed* order so
// fragment loads are contiguous 32B per lane (ds_load_b128) and staging
// stores are single ds_store_b128 per 8-element k-group.
// ---------------------------------------------------------------------------

typedef __attribute__((ext_vector_type(16))) __bf16 v16bf;
typedef __attribute__((ext_vector_type(8)))  float  v8f;

#define NUM_REL 3
#define IN_C    256
#define H1_C    1024
#define H2_C    512
#define H3_C    128
#define OUT_C   21

#define TILE_M 128
#define TILE_N 128
#define TILE_K 32

enum { GF_ACC = 1, GF_RELU = 2 };

// round-to-nearest-even f32 -> bf16, two at a time packed into a dword
__device__ __forceinline__ unsigned pack2bf(float a, float b) {
  unsigned ua = __builtin_bit_cast(unsigned, a);
  ua += 0x7FFFu + ((ua >> 16) & 1u);
  unsigned ub = __builtin_bit_cast(unsigned, b);
  ub += 0x7FFFu + ((ub >> 16) & 1u);
  return (ua >> 16) | (ub & 0xFFFF0000u);
}

// --------------------------- edge-count / norm ------------------------------

__global__ void rgcn_count_kernel(const int* __restrict__ dst,
                                  const int* __restrict__ et,
                                  float* __restrict__ cnt, int E) {
  int e = blockIdx.x * 256 + threadIdx.x;
  if (e < E) {
    atomicAdd(&cnt[(size_t)dst[e] * NUM_REL + et[e]], 1.0f);
  }
}

__global__ void rgcn_norm_kernel(float* __restrict__ cnt, int n) {
  int i = blockIdx.x * 256 + threadIdx.x;
  if (i < n) cnt[i] = 1.0f / fmaxf(cnt[i], 1.0f);
}

// ------------------------------- scatter ------------------------------------
// One block per edge; blockDim.x = in_dim/4 threads, float4 per thread.
// A[(dst*3+et)*in_dim + f] += x[src*in_dim + f] * norm[dst*3+et]

__global__ void rgcn_scatter_kernel(const float* __restrict__ x,
                                    const int* __restrict__ src,
                                    const int* __restrict__ dst,
                                    const int* __restrict__ et,
                                    const float* __restrict__ norm,
                                    float* __restrict__ Abuf, int in_dim) {
  int e = blockIdx.x;
  int s = src[e];
  int d = dst[e];
  int r = et[e];
  size_t seg = (size_t)d * NUM_REL + r;
  float nv = norm[seg];
  int f = threadIdx.x * 4;
  float4 xv = *reinterpret_cast<const float4*>(x + (size_t)s * in_dim + f);
  float* p = Abuf + seg * (size_t)in_dim + f;
  atomicAdd(p + 0, xv.x * nv);
  atomicAdd(p + 1, xv.y * nv);
  atomicAdd(p + 2, xv.z * nv);
  atomicAdd(p + 3, xv.w * nv);
}

// ------------------------- bf16 WMMA GEMM -----------------------------------
// C[M,N] (+)= A[M,K] @ B[K,N]; optional bias + ReLU epilogue.
// 128x128 tile, 256 threads = 8 waves (wave32): wave grid 4(M) x 2(N),
// each wave owns 32x64 = 2x4 fragments of 16x16.
// Requires: N multiple of 128 (grid.x = N/128), K multiple of 32.
//
// LDS holds tiles in fragment-packed order: [frag][lane][16 bf16], so each
// lane's v16bf fragment is a contiguous, 32B-aligned chunk.
//   A elem (m_in_tile, k_in_tile): frag = m>>4, h=(k>>3)&1,
//       lane = (m&15)+16h, idx = k - 8h - (k>=16 ? 8 : 0)
//   B elem (k_in_tile, n_in_tile): frag = n>>4, h = (k>>3)&1,
//       lane = (n&15)+16h, idx = k - 8h - (k>=16 ? 8 : 0)

__global__ void __launch_bounds__(256)
rgcn_gemm_bf16_wmma(const float* __restrict__ A, int lda,
                    const float* __restrict__ B, int ldb,
                    float* __restrict__ C, int ldc,
                    int M, int K,
                    const float* __restrict__ bias, int flags) {
  __shared__ alignas(32) __bf16 AsP[8 * 512];  // 8 frags * 32 lanes * 16 bf16
  __shared__ alignas(32) __bf16 BsP[8 * 512];

  const int blockM = blockIdx.y * TILE_M;
  const int blockN = blockIdx.x * TILE_N;

  const int wave = threadIdx.x >> 5;
  const int lane = threadIdx.x & 31;
  const int half = lane >> 4;
  const int l16  = lane & 15;
  const int wm = (wave & 3) * 32;   // wave row offset in tile
  const int wn = (wave >> 2) * 64;  // wave col offset in tile

  v8f acc[2][4] = {};

  for (int k0 = 0; k0 < K; k0 += TILE_K) {
    // ---- stage A: 512 tasks = (row, 8k-group); 32B global -> one b128 store
#pragma unroll
    for (int t = threadIdx.x; t < 512; t += 256) {
      int r = t >> 2, g = t & 3;       // g: which 8-k group (k = g*8 .. g*8+7)
      int gr = blockM + r;
      float4 v0, v1;
      if (gr < M) {
        const float* ap = A + (size_t)gr * lda + (k0 + g * 8);
        v0 = *reinterpret_cast<const float4*>(ap);
        v1 = *reinterpret_cast<const float4*>(ap + 4);
      } else {
        v0 = make_float4(0.f, 0.f, 0.f, 0.f);
        v1 = v0;
      }
      int h = g & 1;                       // k-group 0,2 -> half 0; 1,3 -> half 1
      int off = ((r >> 4) << 9)            // fragment base
              + (((r & 15) + (h << 4)) << 4)
              + ((g >> 1) << 3);           // idx base 0 or 8
      uint4 p;
      p.x = pack2bf(v0.x, v0.y);
      p.y = pack2bf(v0.z, v0.w);
      p.z = pack2bf(v1.x, v1.y);
      p.w = pack2bf(v1.z, v1.w);
      *reinterpret_cast<uint4*>(&AsP[off]) = p;
    }
    // ---- stage B: 512 tasks = (col, 8k-group); 8 strided loads -> one b128
#pragma unroll
    for (int t = threadIdx.x; t < 512; t += 256) {
      int n = t & 127, g = t >> 7;
      const float* bp = B + (size_t)(k0 + g * 8) * ldb + (blockN + n);
      float f[8];
#pragma unroll
      for (int i = 0; i < 8; ++i) f[i] = bp[(size_t)i * ldb];
      int h = g & 1;                       // k-group 0,2 -> half 0; 1,3 -> half 1
      int off = ((n >> 4) << 9)
              + (((n & 15) + (h << 4)) << 4)
              + ((g >> 1) << 3);           // idx base 0 or 8
      uint4 p;
      p.x = pack2bf(f[0], f[1]);
      p.y = pack2bf(f[2], f[3]);
      p.z = pack2bf(f[4], f[5]);
      p.w = pack2bf(f[6], f[7]);
      *reinterpret_cast<uint4*>(&BsP[off]) = p;
    }
    __syncthreads();

    // ---- fragment loads: contiguous 32B per lane -> ds_load_b128 pairs ----
    v16bf af[2];
#pragma unroll
    for (int fm = 0; fm < 2; ++fm) {
      af[fm] = *reinterpret_cast<const v16bf*>(
          &AsP[(((wm >> 4) + fm) << 9) + (lane << 4)]);
    }
    v16bf bfr[4];
#pragma unroll
    for (int fn = 0; fn < 4; ++fn) {
      bfr[fn] = *reinterpret_cast<const v16bf*>(
          &BsP[(((wn >> 4) + fn) << 9) + (lane << 4)]);
    }

    // ---- 8 WMMAs per wave per K-tile ----
#pragma unroll
    for (int fm = 0; fm < 2; ++fm) {
#pragma unroll
      for (int fn = 0; fn < 4; ++fn) {
        acc[fm][fn] = __builtin_amdgcn_wmma_f32_16x16x32_bf16(
            false, af[fm], false, bfr[fn], (short)0, acc[fm][fn], false, false);
      }
    }
    __syncthreads();
  }

  // --- epilogue: C/D 16x16 f32 layout (VGPR v -> M = v + 8*half, N = l16) ---
#pragma unroll
  for (int fm = 0; fm < 2; ++fm) {
#pragma unroll
    for (int fn = 0; fn < 4; ++fn) {
#pragma unroll
      for (int v = 0; v < 8; ++v) {
        int gr = blockM + wm + fm * 16 + v + 8 * half;
        int gc = blockN + wn + fn * 16 + l16;
        if (gr < M) {
          float out = acc[fm][fn][v];
          size_t idx = (size_t)gr * ldc + gc;
          if (flags & GF_ACC) out += C[idx];
          if (bias) out += bias[gc];
          if (flags & GF_RELU) out = fmaxf(out, 0.0f);
          C[idx] = out;
        }
      }
    }
  }
}

// ----------------------- FC + log_softmax (tiny) ----------------------------

__global__ void __launch_bounds__(128)
rgcn_fc_logsoftmax(const float* __restrict__ h,
                   const float* __restrict__ fcw,
                   const float* __restrict__ fcb,
                   float* __restrict__ out, int Nn) {
  __shared__ float Ws[H3_C * OUT_C];
  __shared__ float Bsm[OUT_C];
  for (int i = threadIdx.x; i < H3_C * OUT_C; i += 128) Ws[i] = fcw[i];
  if (threadIdx.x < OUT_C) Bsm[threadIdx.x] = fcb[threadIdx.x];
  __syncthreads();

  int node = blockIdx.x * 128 + threadIdx.x;
  if (node >= Nn) return;

  float acc[OUT_C];
#pragma unroll
  for (int j = 0; j < OUT_C; ++j) acc[j] = Bsm[j];
  const float* hp = h + (size_t)node * H3_C;
  for (int i = 0; i < H3_C; ++i) {
    float hv = hp[i];
#pragma unroll
    for (int j = 0; j < OUT_C; ++j) acc[j] += hv * Ws[i * OUT_C + j];
  }
  float mx = acc[0];
#pragma unroll
  for (int j = 1; j < OUT_C; ++j) mx = fmaxf(mx, acc[j]);
  float s = 0.0f;
#pragma unroll
  for (int j = 0; j < OUT_C; ++j) s += __expf(acc[j] - mx);
  float lse = mx + __logf(s);
#pragma unroll
  for (int j = 0; j < OUT_C; ++j)
    out[(size_t)node * OUT_C + j] = acc[j] - lse;
}

// ------------------------------- launcher -----------------------------------

static inline void launch_gemm(const float* A, int lda, const float* B, int ldb,
                               float* C, int ldc, int M, int N, int K,
                               const float* bias, int flags, hipStream_t s) {
  dim3 grid(N / TILE_N, (M + TILE_M - 1) / TILE_M);
  rgcn_gemm_bf16_wmma<<<grid, 256, 0, s>>>(A, lda, B, ldb, C, ldc, M, K, bias, flags);
}

extern "C" void kernel_launch(void* const* d_in, const int* in_sizes, int n_in,
                              void* d_out, int out_size, void* d_ws, size_t ws_size,
                              hipStream_t stream) {
  const float* x     = (const float*)d_in[0];
  const int*   eidx  = (const int*)d_in[1];
  const int*   etp   = (const int*)d_in[2];
  const float* W1    = (const float*)d_in[3];
  const float* root1 = (const float*)d_in[4];
  const float* b1    = (const float*)d_in[5];
  const float* W2    = (const float*)d_in[6];
  const float* root2 = (const float*)d_in[7];
  const float* b2    = (const float*)d_in[8];
  const float* W3    = (const float*)d_in[9];
  const float* root3 = (const float*)d_in[10];
  const float* b3    = (const float*)d_in[11];
  const float* fcw   = (const float*)d_in[12];
  const float* fcb   = (const float*)d_in[13];

  const int Nn = in_sizes[0] / IN_C;   // 40000
  const int E  = in_sizes[2];          // 500000
  const int* srcp = eidx;
  const int* dstp = eidx + E;

  // workspace layout (f32): norm | Abuf(max N*3*1024) | h1 | h2 | h3
  char* wsb = (char*)d_ws;
  size_t o = 0;
  float* normb = (float*)(wsb + o);
  o += (((size_t)Nn * NUM_REL * 4) + 511) & ~(size_t)511;
  float* Abuf = (float*)(wsb + o);
  o += (size_t)Nn * NUM_REL * H1_C * 4;   // 491.5 MB (max in_dim = 1024)
  float* h1 = (float*)(wsb + o); o += (size_t)Nn * H1_C * 4;
  float* h2 = (float*)(wsb + o); o += (size_t)Nn * H2_C * 4;
  float* h3 = (float*)(wsb + o); o += (size_t)Nn * H3_C * 4;

  // ---- edge norms: count per (dst, rel), invert in place ----
  hipMemsetAsync(normb, 0, (size_t)Nn * NUM_REL * 4, stream);
  rgcn_count_kernel<<<(E + 255) / 256, 256, 0, stream>>>(dstp, etp, normb, E);
  rgcn_norm_kernel<<<(Nn * NUM_REL + 255) / 256, 256, 0, stream>>>(normb, Nn * NUM_REL);

  // ---- layer 1: 256 -> 1024 ----
  hipMemsetAsync(Abuf, 0, (size_t)Nn * NUM_REL * IN_C * 4, stream);
  rgcn_scatter_kernel<<<E, IN_C / 4, 0, stream>>>(x, srcp, dstp, etp, normb, Abuf, IN_C);
  launch_gemm(Abuf, NUM_REL * IN_C, W1, H1_C, h1, H1_C,
              Nn, H1_C, NUM_REL * IN_C, nullptr, 0, stream);
  launch_gemm(x, IN_C, root1, H1_C, h1, H1_C,
              Nn, H1_C, IN_C, b1, GF_ACC | GF_RELU, stream);

  // ---- layer 2: 1024 -> 512 ----
  hipMemsetAsync(Abuf, 0, (size_t)Nn * NUM_REL * H1_C * 4, stream);
  rgcn_scatter_kernel<<<E, H1_C / 4, 0, stream>>>(h1, srcp, dstp, etp, normb, Abuf, H1_C);
  launch_gemm(Abuf, NUM_REL * H1_C, W2, H2_C, h2, H2_C,
              Nn, H2_C, NUM_REL * H1_C, nullptr, 0, stream);
  launch_gemm(h1, H1_C, root2, H2_C, h2, H2_C,
              Nn, H2_C, H1_C, b2, GF_ACC | GF_RELU, stream);

  // ---- layer 3: 512 -> 128 ----
  hipMemsetAsync(Abuf, 0, (size_t)Nn * NUM_REL * H2_C * 4, stream);
  rgcn_scatter_kernel<<<E, H2_C / 4, 0, stream>>>(h2, srcp, dstp, etp, normb, Abuf, H2_C);
  launch_gemm(Abuf, NUM_REL * H2_C, W3, H3_C, h3, H3_C,
              Nn, H3_C, NUM_REL * H2_C, nullptr, 0, stream);
  launch_gemm(h2, H2_C, root3, H3_C, h3, H3_C,
              Nn, H3_C, H2_C, b3, GF_ACC | GF_RELU, stream);

  // ---- FC + log_softmax ----
  rgcn_fc_logsoftmax<<<(Nn + 127) / 128, 128, 0, stream>>>(h3, fcw, fcb, (float*)d_out, Nn);
}
